// GQASelfAttentionRelPos_47433618817470
// MI455X (gfx1250) — compile-verified, matrix operand order checked by reading, not run on hardware
//
#include <hip/hip_runtime.h>
#include <hip/hip_bf16.h>

typedef __bf16 bf16_t;
typedef __attribute__((ext_vector_type(16))) __bf16 v16bf;
typedef __attribute__((ext_vector_type(8)))  float  v8f;
typedef unsigned int u32x4 __attribute__((ext_vector_type(4), may_alias));
typedef __attribute__((ext_vector_type(4)))  float  f32x4;
typedef __attribute__((ext_vector_type(4)))  __bf16 v4bf;

union FragU { u32x4 u[2]; v16bf v; };

__device__ __forceinline__ v8f wmma_bf16(v16bf a, v16bf b, v8f c) {
  // D = A(16x32 bf16) x B(32x16 bf16) + C(16x16 f32)
  return __builtin_amdgcn_wmma_f32_16x16x32_bf16(false, a, false, b, (short)0, c,
                                                 false, false);
}

// A/B fragment (16x32 bf16) from LDS (row-major, `stride` halves/row), K base `koff`.
// lane<16: row=lane, K in [0,8)+[16,24); lane>=16: row=lane-16, K in [8,16)+[24,32)
__device__ __forceinline__ v16bf load_frag_lds(const bf16_t* base, int lane,
                                               int stride, int koff) {
  const int row = lane & 15;
  const int c   = ((lane < 16) ? 0 : 8) + koff;
  FragU f;
  f.u[0] = *(const u32x4*)(base + row * stride + c);
  f.u[1] = *(const u32x4*)(base + row * stride + c + 16);
  return f.v;
}

// Same fragment from a per-lane global row pointer (row already selected by caller)
__device__ __forceinline__ v16bf load_frag_g(const bf16_t* rowptr, int lane, int koff) {
  const int c = ((lane < 16) ? 0 : 8) + koff;
  FragU f;
  f.u[0] = *(const u32x4*)(rowptr + c);
  f.u[1] = *(const u32x4*)(rowptr + c + 16);
  return f.v;
}

// CDNA5 async DMA: one 16B chunk global -> LDS, tracked by ASYNCcnt (no VGPR transit)
__device__ __forceinline__ void async_copy_b128(const bf16_t* lds_dst, const bf16_t* gsrc) {
  const unsigned l = (unsigned)(unsigned long long)lds_dst;  // low 32b of LDS aperture addr
  const unsigned long long g = (unsigned long long)gsrc;
  asm volatile("global_load_async_to_lds_b128 %0, %1, off" :: "v"(l), "v"(g) : "memory");
}
__device__ __forceinline__ void async_wait0() {
  asm volatile("s_wait_asynccnt 0" ::: "memory");
}

// ---------------- conversion kernels ----------------
__global__ void cvt_f32_to_bf16_v4(const float* __restrict__ in,
                                   bf16_t* __restrict__ out, int n4) {
  int i = blockIdx.x * 256 + threadIdx.x;
  if (i < n4) {
    f32x4 f = ((const f32x4*)in)[i];
    v4bf o;
    o.x = (bf16_t)f.x; o.y = (bf16_t)f.y; o.z = (bf16_t)f.z; o.w = (bf16_t)f.w;
    ((v4bf*)out)[i] = o;
  }
}

// E: [255,64] f32 -> Eh: [256,64] bf16 (row 255 zero pad)
__global__ void cvt_E_pad(const float* __restrict__ E, bf16_t* __restrict__ Eh) {
  int i = blockIdx.x * 256 + threadIdx.x;
  if (i < 256 * 64) Eh[i] = (i < 255 * 64) ? (bf16_t)E[i] : (bf16_t)0.0f;
}

// ---------------- generic WMMA GEMM:  C[m,n] = sum_k A[m,k] * W[n,k] ----------------
// MODE 0: bf16 row-major (ldo), scaled          MODE 1: q scatter -> [B,H,T,64] bf16
// MODE 2: kv scatter -> [B,G,T,64] bf16          MODE 3: f32 row-major (ldo)
template <int MODE>
__device__ __forceinline__ void store_tile(void* out, v8f c, int m0, int n0,
                                           int lane, float scale, int ldo) {
  const int col   = lane & 15;
  const int rbase = (lane < 16) ? 0 : 8;
#pragma unroll
  for (int i = 0; i < 8; ++i) {
    const int m = m0 + rbase + i;
    const int n = n0 + col;
    const float v = c[i] * scale;
    if (MODE == 0) {
      ((bf16_t*)out)[(size_t)m * ldo + n] = (bf16_t)v;
    } else if (MODE == 1) {
      const int b = m >> 10, t = m & 1023, h = n >> 6, d = n & 63;
      ((bf16_t*)out)[(((size_t)(b * 16 + h) * 1024) + t) * 64 + d] = (bf16_t)v;
    } else if (MODE == 2) {
      const int b = m >> 10, t = m & 1023, g = n >> 6, d = n & 63;
      ((bf16_t*)out)[(((size_t)(b * 4 + g) * 1024) + t) * 64 + d] = (bf16_t)v;
    } else {
      ((float*)out)[(size_t)m * ldo + n] = v;
    }
  }
}

template <int MODE>
__launch_bounds__(256)
__global__ void gemm_bf16_wmma(const bf16_t* __restrict__ A,
                               const bf16_t* __restrict__ W,
                               void* __restrict__ out,
                               int M, int N, int K, float scale, int ldo) {
  __shared__ bf16_t sA[2][128 * 40];
  __shared__ bf16_t sW[2][64 * 40];
  const int tid  = threadIdx.x;
  const int lane = tid & 31;
  const int wave = tid >> 5;
  const int wm   = (wave & 3) * 32;
  const int wn   = (wave >> 2) * 32;
  const int m0   = blockIdx.y * 128;
  const int n0   = blockIdx.x * 64;

  // per-thread staging coordinates (8-half chunks)
  const int rA0 = tid >> 2, cc = (tid & 3) * 8;
  const int rA1 = rA0 + 64;
  const int rW  = tid >> 2;

  v8f c00 = {}, c01 = {}, c10 = {}, c11 = {};

  // prologue: async-stage k-tile 0 into buffer 0
  async_copy_b128(&sA[0][rA0 * 40 + cc], &A[(size_t)(m0 + rA0) * K + cc]);
  async_copy_b128(&sA[0][rA1 * 40 + cc], &A[(size_t)(m0 + rA1) * K + cc]);
  async_copy_b128(&sW[0][rW * 40 + cc],  &W[(size_t)(n0 + rW) * K + cc]);

  int buf = 0;
  for (int kk = 0; kk < K; kk += 32, buf ^= 1) {
    async_wait0();
    __syncthreads();
    if (kk + 32 < K) {  // prefetch next k-tile into the other buffer (overlaps WMMA)
      const int kn = kk + 32;
      async_copy_b128(&sA[buf ^ 1][rA0 * 40 + cc], &A[(size_t)(m0 + rA0) * K + kn + cc]);
      async_copy_b128(&sA[buf ^ 1][rA1 * 40 + cc], &A[(size_t)(m0 + rA1) * K + kn + cc]);
      async_copy_b128(&sW[buf ^ 1][rW * 40 + cc],  &W[(size_t)(n0 + rW) * K + kn + cc]);
    }
    v16bf a0 = load_frag_lds(&sA[buf][wm * 40],        lane, 40, 0);
    v16bf a1 = load_frag_lds(&sA[buf][(wm + 16) * 40], lane, 40, 0);
    v16bf b0 = load_frag_lds(&sW[buf][wn * 40],        lane, 40, 0);
    v16bf b1 = load_frag_lds(&sW[buf][(wn + 16) * 40], lane, 40, 0);
    c00 = wmma_bf16(a0, b0, c00);
    c01 = wmma_bf16(a0, b1, c01);
    c10 = wmma_bf16(a1, b0, c10);
    c11 = wmma_bf16(a1, b1, c11);
  }

  store_tile<MODE>(out, c00, m0 + wm,      n0 + wn,      lane, scale, ldo);
  store_tile<MODE>(out, c01, m0 + wm,      n0 + wn + 16, lane, scale, ldo);
  store_tile<MODE>(out, c10, m0 + wm + 16, n0 + wn,      lane, scale, ldo);
  store_tile<MODE>(out, c11, m0 + wm + 16, n0 + wn + 16, lane, scale, ldo);
}

// ---------------- flash attention + rel-pos bias ----------------
// Q: [B,H,T,64] bf16 (pre-scaled); K,V: [B,G,T,64] bf16; P: [B,H,T,256] bf16 bias
// O: [B,T,D] bf16.  Block = 8 waves x 16 query rows; 64-key chunks; softmax
// denominator accumulated by WMMA via a ones-row appended to V^T.
__launch_bounds__(256)
__global__ void flash_attn_relpos(const bf16_t* __restrict__ Q,
                                  const bf16_t* __restrict__ Kb,
                                  const bf16_t* __restrict__ Vb,
                                  const bf16_t* __restrict__ P,
                                  bf16_t* __restrict__ O) {
  __shared__ bf16_t sPP[128 * 256];   // bias rows for this block's 128 queries (64KB)
  __shared__ bf16_t sVt[80 * 72];     // V^T chunk [dim][key] + ones row at dim 64
  __shared__ bf16_t sPr[8][16 * 72];  // per-wave probs 16x64

  const int tid  = threadIdx.x;
  const int lane = tid & 31;
  const int wave = tid >> 5;
  const int b = blockIdx.z, h = blockIdx.y, g = h >> 2;
  const int qblk  = blockIdx.x * 128;
  const int qbase = qblk + wave * 16;
  const int rsel  = (lane < 16) ? 0 : 8;
  const int col   = lane & 15;

  // async-stage the 128x256 bias table slice for this block
  {
    const bf16_t* Psrc = P + ((size_t)(b * 16 + h) * 1024 + qblk) * 256;
#pragma unroll
    for (int j = 0; j < 16; ++j) {
      const int off = (tid * 16 + j) * 8;
      async_copy_b128(&sPP[off], Psrc + off);
    }
  }
  if (tid < 64) sVt[64 * 72 + tid] = (bf16_t)1.0f;  // ones row (written once)

  const bf16_t* qrow = Q + (((size_t)(b * 16 + h) * 1024) + (qbase + col)) * 64;
  const v16bf qa0 = load_frag_g(qrow, lane, 0);
  const v16bf qa1 = load_frag_g(qrow, lane, 32);

  const v8f zero = {};
  v8f o[5];
  float mrow[8];
#pragma unroll
  for (int nb = 0; nb < 5; ++nb) o[nb] = zero;
#pragma unroll
  for (int i = 0; i < 8; ++i) mrow[i] = -1e30f;

  const bf16_t* kbase = Kb + ((size_t)(b * 4 + g) * 1024) * 64;
  const bf16_t* vbase = Vb + ((size_t)(b * 4 + g) * 1024) * 64;

  async_wait0();
  __syncthreads();  // sPP + ones row visible

  const int vkey = tid >> 2;
  const int vd0  = (tid & 3) * 16;

  for (int kt = 0; kt < 1024; kt += 64) {
    __syncthreads();  // previous chunk's sVt reads finished
    {  // stage V chunk transposed: (64 keys x 64 dims) -> sVt[dim][key]
      union { u32x4 u; bf16_t hh[8]; } c0, c1;
      const bf16_t* vr = vbase + (size_t)(kt + vkey) * 64 + vd0;
      c0.u = *(const u32x4*)vr;
      c1.u = *(const u32x4*)(vr + 8);
#pragma unroll
      for (int j = 0; j < 8; ++j) {
        sVt[(vd0 + j) * 72 + vkey]     = c0.hh[j];
        sVt[(vd0 + 8 + j) * 72 + vkey] = c1.hh[j];
      }
    }
    __syncthreads();

    // scores: 4 x (16q x 16k) tiles, K fragments straight from global
    v8f s[4];
#pragma unroll
    for (int kk = 0; kk < 4; ++kk) {
      const bf16_t* krow = kbase + (size_t)(kt + kk * 16 + col) * 64;
      v8f acc = zero;
      acc = wmma_bf16(qa0, load_frag_g(krow, lane, 0),  acc);
      acc = wmma_bf16(qa1, load_frag_g(krow, lane, 32), acc);
      s[kk] = acc;
    }

    // bias (LDS gather) + online softmax (max-tree only; sum comes from WMMA)
#pragma unroll
    for (int i = 0; i < 8; ++i) {
      const int r  = rsel + i;
      const int qg = qbase + r;
      const bf16_t* pp = &sPP[(wave * 16 + r) * 256];
      float t[4];
#pragma unroll
      for (int kk = 0; kk < 4; ++kk) {
        int d = qg - (kt + kk * 16 + col);
        d = d < -127 ? -127 : (d > 127 ? 127 : d);
        t[kk] = s[kk][i] + (float)pp[d + 127];
      }
      float mx = fmaxf(fmaxf(t[0], t[1]), fmaxf(t[2], t[3]));
#pragma unroll
      for (int off2 = 8; off2 > 0; off2 >>= 1) mx = fmaxf(mx, __shfl_xor(mx, off2, 32));
      const float mnew  = fmaxf(mrow[i], mx);
      const float alpha = __expf(mrow[i] - mnew);
      mrow[i] = mnew;
#pragma unroll
      for (int nb = 0; nb < 5; ++nb) o[nb][i] *= alpha;
      bf16_t* pr = &sPr[wave][r * 72];
#pragma unroll
      for (int kk = 0; kk < 4; ++kk)
        pr[kk * 16 + col] = (bf16_t)__expf(t[kk] - mnew);
    }
    asm volatile("" ::: "memory");  // keep LDS prob stores before fragment reads

    const v16bf pa0 = load_frag_lds(&sPr[wave][0], lane, 72, 0);
    const v16bf pa1 = load_frag_lds(&sPr[wave][0], lane, 72, 32);
#pragma unroll
    for (int nb = 0; nb < 5; ++nb) {  // nb==4 accumulates the softmax denominator
      o[nb] = wmma_bf16(pa0, load_frag_lds(&sVt[nb * 16 * 72], lane, 72, 0),  o[nb]);
      o[nb] = wmma_bf16(pa1, load_frag_lds(&sVt[nb * 16 * 72], lane, 72, 32), o[nb]);
    }
  }

#pragma unroll
  for (int i = 0; i < 8; ++i) {
    const int row = qbase + rsel + i;
    const float lsum = __shfl(o[4][i], lane & 16, 32);  // ones-column (N==0) holds l
    const float inv  = 1.0f / lsum;
    bf16_t* orow = O + ((size_t)b * 1024 + row) * 1024 + h * 64;
    orow[col]      = (bf16_t)(o[0][i] * inv);
    orow[col + 16] = (bf16_t)(o[1][i] * inv);
    orow[col + 32] = (bf16_t)(o[2][i] * inv);
    orow[col + 48] = (bf16_t)(o[3][i] * inv);
  }
}

// ---------------- host launch ----------------
extern "C" void kernel_launch(void* const* d_in, const int* in_sizes, int n_in,
                              void* d_out, int out_size, void* d_ws, size_t ws_size,
                              hipStream_t stream) {
  (void)in_sizes; (void)n_in; (void)out_size; (void)ws_size;
  const float* x  = (const float*)d_in[0];
  const float* Wq = (const float*)d_in[1];
  const float* Wk = (const float*)d_in[2];
  const float* Wv = (const float*)d_in[3];
  const float* Wo = (const float*)d_in[4];
  const float* E  = (const float*)d_in[5];

  char* ws = (char*)d_ws;
  size_t off = 0;
  auto alloc = [&](size_t bytes) {
    char* p = ws + off;
    off += (bytes + 255) & ~(size_t)255;
    return p;
  };
  bf16_t* xh  = (bf16_t*)alloc(8ull * 1024 * 1024 * 2);
  bf16_t* Wqh = (bf16_t*)alloc(1024ull * 1024 * 2);
  bf16_t* Wkh = (bf16_t*)alloc(256ull * 1024 * 2);
  bf16_t* Wvh = (bf16_t*)alloc(256ull * 1024 * 2);
  bf16_t* Woh = (bf16_t*)alloc(1024ull * 1024 * 2);
  bf16_t* Eh  = (bf16_t*)alloc(256ull * 64 * 2);
  bf16_t* qh  = (bf16_t*)alloc(8ull * 16 * 1024 * 64 * 2);   // [B,H,T,64]
  bf16_t* kh  = (bf16_t*)alloc(8ull * 4 * 1024 * 64 * 2);    // [B,G,T,64]
  bf16_t* vh  = (bf16_t*)alloc(8ull * 4 * 1024 * 64 * 2);
  bf16_t* Pb  = (bf16_t*)alloc(8ull * 16 * 1024 * 256 * 2);  // bias table
  bf16_t* ao  = (bf16_t*)alloc(8ull * 1024 * 1024 * 2);      // attn out

  cvt_f32_to_bf16_v4<<<(2097152 + 255) / 256, 256, 0, stream>>>(x,  xh,  2097152);
  cvt_f32_to_bf16_v4<<<(262144  + 255) / 256, 256, 0, stream>>>(Wq, Wqh, 262144);
  cvt_f32_to_bf16_v4<<<(65536   + 255) / 256, 256, 0, stream>>>(Wk, Wkh, 65536);
  cvt_f32_to_bf16_v4<<<(65536   + 255) / 256, 256, 0, stream>>>(Wv, Wvh, 65536);
  cvt_f32_to_bf16_v4<<<(262144  + 255) / 256, 256, 0, stream>>>(Wo, Woh, 262144);
  cvt_E_pad<<<64, 256, 0, stream>>>(E, Eh);

  // projections (y = x @ W.T); q fused with 1/sqrt(64) scale + [B,H,T,64] scatter
  gemm_bf16_wmma<1><<<dim3(16, 64),  256, 0, stream>>>(xh, Wqh, qh, 8192, 1024, 1024, 0.125f, 0);
  gemm_bf16_wmma<2><<<dim3(4, 64),   256, 0, stream>>>(xh, Wkh, kh, 8192, 256, 1024, 1.0f, 0);
  gemm_bf16_wmma<2><<<dim3(4, 64),   256, 0, stream>>>(xh, Wvh, vh, 8192, 256, 1024, 1.0f, 0);

  // rel-pos bias table: P[b,h,q,j] = q_scaled . E[j]
  gemm_bf16_wmma<0><<<dim3(4, 1024), 256, 0, stream>>>(qh, Eh, Pb, 131072, 256, 64, 1.0f, 256);

  flash_attn_relpos<<<dim3(8, 16, 8), 256, 0, stream>>>(qh, kh, vh, Pb, ao);

  // output projection -> f32 d_out
  gemm_bf16_wmma<3><<<dim3(16, 64), 256, 0, stream>>>(ao, Woh, d_out, 8192, 1024, 1024, 1.0f, 1024);
}
